// MyRNN_46239617908955
// MI455X (gfx1250) — compile-verified, matrix operand order checked by reading, not run on hardware
//
#include <hip/hip_runtime.h>
#include <math.h>

typedef __attribute__((ext_vector_type(16))) _Float16 v16h;
typedef __attribute__((ext_vector_type(8)))  float    v8f;
typedef __attribute__((ext_vector_type(4)))  float    v4f;

#define BATCH   512
#define MAXLEN  512
#define EMBED   256
#define UNITS   32
#define F1DIM   16
#define ROWS    (BATCH * MAXLEN)
#define TILES   (ROWS / 16)

// ---------------------------------------------------------------------------
// Phase 1: U[r, 0:32] = emb[tokens[r], 0:256] @ Wx + bias   (r = b*T + t)
// One wave computes a 16x32 tile of U via 2 chained v_wmma_f32_16x16x32_f16
// per K-step (8 K-steps for K=256). Wx is staged in LDS in the exact
// B-operand lane layout so the hot loop loads it as one 32-byte vector.
// ---------------------------------------------------------------------------
__global__ void __launch_bounds__(128) rnn_phase1_emb_gemm(
    const int*   __restrict__ tokens,
    const float* __restrict__ emb,
    const float* __restrict__ Wx,
    const float* __restrict__ bias,
    float*       __restrict__ U)
{
    // [kstep 0..7][ntile 0..1][lane 0..31][16 halfs]  = 16 KB
    __shared__ __align__(32) _Float16 Blds[8 * 2 * 32 * 16];

    const int tid = threadIdx.x;

    // Stage Wx (f32 [256,32]) into LDS, pre-swizzled into the WMMA B layout:
    // lane n holds column N = ntile*16 + (n&15); lanes>=16 take the K+8 half.
    for (int slot = tid; slot < 8 * 2 * 32; slot += 128) {
        const int ks   = slot >> 6;        // k-step 0..7
        const int nt   = (slot >> 5) & 1;  // n-tile 0..1
        const int ln   = slot & 31;        // lane slot 0..31
        const int ncol = nt * 16 + (ln & 15);
        const int kadd = (ln >= 16) ? 8 : 0;
        _Float16* dst = &Blds[slot * 16];
        #pragma unroll
        for (int j = 0; j < 16; ++j) {
            const int k = ks * 32 + kadd + ((j < 8) ? j : (j + 8));
            dst[j] = (_Float16)Wx[k * UNITS + ncol];
        }
    }
    __syncthreads();

    const int lane  = tid & 31;
    const int wave  = tid >> 5;
    const int m     = lane & 15;               // M row within tile (A layout)
    const int khalf = (lane >= 16) ? 8 : 0;    // K offset for this lane half
    const float bias0 = bias[m];
    const float bias1 = bias[16 + m];

    const int wavesInGrid = gridDim.x * 4;
    const int waveId      = blockIdx.x * 4 + wave;

    for (int rt = waveId; rt < TILES; rt += wavesInGrid) {
        const int rBase = rt * 16;
        const int row   = rBase + m;
        const int token = tokens[row];
        const float* erow = emb + (size_t)token * EMBED;

        v8f acc0 = {};
        v8f acc1 = {};
        #pragma unroll
        for (int ks = 0; ks < 8; ++ks) {
            const int kb = ks * 32 + khalf;
            // A layout: halfs 0..7  -> K = kb+{0..7}
            //           halfs 8..15 -> K = kb+{16..23}   (shifted by khalf)
            const v4f f0 = *(const v4f*)(erow + kb);
            const v4f f1 = *(const v4f*)(erow + kb + 4);
            const v4f f2 = *(const v4f*)(erow + kb + 16);
            const v4f f3 = *(const v4f*)(erow + kb + 20);
            v16h a;
            #pragma unroll
            for (int j = 0; j < 4; ++j) {
                a[j]      = (_Float16)f0[j];
                a[j + 4]  = (_Float16)f1[j];
                a[j + 8]  = (_Float16)f2[j];
                a[j + 12] = (_Float16)f3[j];
            }
            const v16h b0 = *(const v16h*)&Blds[((ks * 2 + 0) * 32 + lane) * 16];
            const v16h b1 = *(const v16h*)&Blds[((ks * 2 + 1) * 32 + lane) * 16];
            acc0 = __builtin_amdgcn_wmma_f32_16x16x32_f16(
                       false, a, false, b0, (short)0, acc0, false, false);
            acc1 = __builtin_amdgcn_wmma_f32_16x16x32_f16(
                       false, a, false, b1, (short)0, acc1, false, false);
        }

        // D layout: lane<16 -> M=v, N=lane ; lane>=16 -> M=v+8, N=lane-16
        const int mb = (lane >= 16) ? 8 : 0;
        #pragma unroll
        for (int v = 0; v < 8; ++v) {
            float* urow = U + (size_t)(rBase + mb + v) * UNITS;
            urow[m]      = acc0[v] + bias0;
            urow[16 + m] = acc1[v] + bias1;
        }
    }
}

// ---------------------------------------------------------------------------
// Phase 2: sequential tanh recurrence + dense head, one wave per batch row.
// lane u holds h[u]; Wh column u lives in registers; h[k] broadcast via shfl.
// ---------------------------------------------------------------------------
__global__ void __launch_bounds__(128) rnn_phase2_scan(
    const float* __restrict__ U,
    const float* __restrict__ Wh,
    const float* __restrict__ W1,
    const float* __restrict__ b1,
    const float* __restrict__ W2,
    const float* __restrict__ b2,
    float*       __restrict__ out)
{
    const int lane = threadIdx.x & 31;
    const int b    = blockIdx.x * 4 + (threadIdx.x >> 5);
    if (b >= BATCH) return;

    float whcol[UNITS];
    #pragma unroll
    for (int k = 0; k < UNITS; ++k) whcol[k] = Wh[k * UNITS + lane];

    float h = 0.0f;
    const float* ub = U + (size_t)b * MAXLEN * UNITS;

    for (int t = 0; t < MAXLEN; ++t) {
        float acc = ub[t * UNITS + lane];   // x_t@Wx + bias, precomputed
        #pragma unroll
        for (int k = 0; k < UNITS; ++k)
            acc = fmaf(__shfl(h, k, 32), whcol[k], acc);
        h = tanhf(acc);
    }

    // Head: f = h @ W1 + b1 (16 wide), z = f @ W2 + b2, sigmoid.
    float f = (lane < F1DIM) ? b1[lane] : 0.0f;
    #pragma unroll
    for (int k = 0; k < UNITS; ++k) {
        const float hk = __shfl(h, k, 32);
        if (lane < F1DIM) f = fmaf(hk, W1[k * F1DIM + lane], f);
    }
    float val = (lane < F1DIM) ? f * W2[lane] : 0.0f;
    #pragma unroll
    for (int off = 16; off > 0; off >>= 1)
        val += __shfl_down(val, off, 32);
    if (lane == 0) {
        const float z = val + b2[0];
        out[b] = 1.0f / (1.0f + expf(-z));
    }
}

// ---------------------------------------------------------------------------
extern "C" void kernel_launch(void* const* d_in, const int* in_sizes, int n_in,
                              void* d_out, int out_size, void* d_ws, size_t ws_size,
                              hipStream_t stream)
{
    (void)in_sizes; (void)n_in; (void)out_size; (void)ws_size;
    const int*   tokens = (const int*)  d_in[0];
    const float* emb    = (const float*)d_in[1];
    const float* Wx     = (const float*)d_in[2];
    const float* Wh     = (const float*)d_in[3];
    const float* bvec   = (const float*)d_in[4];
    const float* W1     = (const float*)d_in[5];
    const float* b1     = (const float*)d_in[6];
    const float* W2     = (const float*)d_in[7];
    const float* b2     = (const float*)d_in[8];
    float* out = (float*)d_out;
    float* U   = (float*)d_ws;   // ROWS * 32 * 4 B = 32 MB scratch

    rnn_phase1_emb_gemm<<<1024, 128, 0, stream>>>(tokens, emb, Wx, bvec, U);
    rnn_phase2_scan<<<(BATCH + 3) / 4, 128, 0, stream>>>(U, Wh, W1, b1, W2, b2, out);
}
